// VoxelConvolution_22093311771093
// MI455X (gfx1250) — compile-verified, matrix operand order checked by reading, not run on hardware
//
#include <hip/hip_runtime.h>

// ---- CDNA5 vector types for WMMA ----
typedef __attribute__((ext_vector_type(16))) __bf16       v16bf;
typedef __attribute__((ext_vector_type(8)))  float        v8f;
typedef __attribute__((ext_vector_type(4)))  unsigned int uint4v;
typedef __attribute__((ext_vector_type(4)))  float        float4v;

#define C_MUL 48
#define V_MUL 24
#define DIM   120          // in_dim == out_dim
#define NLAT  125          // 5x5x5 taps
#define LDSW  136          // padded LDS row stride (bf16 elems): 272B -> bank-staggered b128 reads

#define KBUF_BYTES  4096000ull                 // 125*4*8*32*16 bf16 fragments
#define XBF_ELEMS   53084160ull                // 4*48*48*48*120
#define XBF_BYTES   (XBF_ELEMS * 2ull)
#define ZP_OFF      (KBUF_BYTES + XBF_BYTES)
#define WS_NEED     (ZP_OFF + 256ull)

// round-to-nearest bf16 (packs two)
__device__ __forceinline__ unsigned int bf16pair(float a, float b) {
    unsigned int ua = __float_as_uint(a); ua += 0x7FFFu + ((ua >> 16) & 1u);
    unsigned int ub = __float_as_uint(b); ub += 0x7FFFu + ((ub >> 16) & 1u);
    return (ua >> 16) | (ub & 0xFFFF0000u);
}
__device__ __forceinline__ unsigned short bf16r(float a) {
    unsigned int ua = __float_as_uint(a); ua += 0x7FFFu + ((ua >> 16) & 1u);
    return (unsigned short)(ua >> 16);
}
// soft unit step: x>0 ? exp(-1/x) : 0
__device__ __forceinline__ float sus(float x) { return x > 0.f ? __expf(-1.f / x) : 0.f; }

// -----------------------------------------------------------------------------
// Phase 1: synthesize conv kernel into WMMA B-fragment order, bf16.
// Layout: Bbuf[(((t*4 + kt)*8 + nt)*32 + lane)*16 + j]
//   lane<16 : o = nt*16+lane,    K = kt*32     + (j<8 ? j : j+8)
//   lane>=16: o = nt*16+lane-16, K = kt*32 + 8 + (j<8 ? j : j+8)
// Self-connection folded into center tap t==62.
// -----------------------------------------------------------------------------
__global__ __launch_bounds__(256) void build_wmma_kernel(
    const float* __restrict__ weight,      // (3, 5184)
    const float* __restrict__ sc_weight,   // (2880,)
    unsigned short* __restrict__ Bbuf)     // 4 MB
{
    int idx = blockIdx.x * 256 + threadIdx.x;
    if (idx >= NLAT * 16384) return;
    int j    = idx & 15;
    int lane = (idx >> 4) & 31;
    int nt   = (idx >> 9) & 7;
    int kt   = (idx >> 12) & 3;
    int t    = idx >> 14;

    int o = nt * 16 + (lane & 15);
    int i = kt * 32 + ((lane >> 4) << 3) + (j < 8 ? j : j + 8);

    float val = 0.f;
    if (i < DIM && o < DIM) {
        float d0 = (float)(t / 25) - 2.f;
        float d1 = (float)((t / 5) % 5) - 2.f;
        float d2 = (float)(t % 5) - 2.f;
        float nrm = sqrtf(d0*d0 + d1*d1 + d2*d2);
        float df;
        df = (nrm - 0.625f) * 1.6f; float emb0 = 8.433573f * sus(df + 1.f) * sus(1.f - df);
        df = (nrm - 1.25f ) * 1.6f; float emb1 = 8.433573f * sus(df + 1.f) * sus(1.f - df);
        df = (nrm - 1.875f) * 1.6f; float emb2 = 8.433573f * sus(df + 1.f) * sus(1.f - df);
        float inv = nrm > 0.f ? (1.7320508f / nrm) : 0.f;
        float sh1x = d0 * inv, sh1y = d1 * inv, sh1z = d2 * inv;

        const float A0  = 0.11785113f;  // 1/sqrt(72)
        const float A1  = 0.20412415f;  // sqrt(3)/sqrt(72)
        const float IS3 = 0.57735027f;  // 1/sqrt(3)
        const float RN  = 1.f / 125.f;

        int n = -1; float sc = 0.f;
        if (i < C_MUL && o < C_MUL) {                       // scalar->scalar
            n = i * C_MUL + o;
            sc = A0 * RN;
        } else if (i < C_MUL) {                             // scalar->vector
            int c = o - C_MUL, v = c / 3, jj = c % 3;
            n = C_MUL*C_MUL + i * V_MUL + v;
            float sh = (jj == 0) ? sh1x : (jj == 1) ? sh1y : sh1z;
            sc = A1 * IS3 * RN * sh;
        } else if (o >= C_MUL) {                            // vector->vector
            int r = i - C_MUL, c = o - C_MUL;
            int u = r / 3, a = r % 3, v = c / 3, bb = c % 3;
            if (a == bb) { n = C_MUL*C_MUL + C_MUL*V_MUL + u * V_MUL + v; sc = A1 * IS3 * RN; }
        } else {                                            // vector->scalar
            int r = i - C_MUL, u = r / 3, a = r % 3;
            n = C_MUL*C_MUL + C_MUL*V_MUL + V_MUL*V_MUL + u * C_MUL + o;
            float sh = (a == 0) ? sh1x : (a == 1) ? sh1y : sh1z;
            sc = A0 * IS3 * RN * sh;
        }
        if (n >= 0) {
            float w = emb0 * weight[n] + emb1 * weight[5184 + n] + emb2 * weight[2 * 5184 + n];
            val = sc * w;
        }
        if (t == 62) {  // center tap: fold the self-connection (1x1 channel mix)
            if (i < C_MUL && o < C_MUL) {
                val += sc_weight[i * C_MUL + o] * 0.14433757f;          // 1/sqrt(48)
            } else if (i >= C_MUL && o >= C_MUL) {
                int r = i - C_MUL, c = o - C_MUL;
                int v = r / 3, bb = r % 3, u = c / 3, a = c % 3;
                if (a == bb)
                    val += sc_weight[C_MUL*C_MUL + v * V_MUL + u] * 0.20412415f;  // 1/sqrt(24)
            }
        }
    }
    Bbuf[idx] = bf16r(val);
}

// -----------------------------------------------------------------------------
// Phase 1b: pre-convert x to bf16 (one-shot, bandwidth-trivial) + zero page.
// -----------------------------------------------------------------------------
__global__ __launch_bounds__(256) void convert_bf16(
    const float* __restrict__ x,
    unsigned short* __restrict__ xbf,
    unsigned short* __restrict__ zp)
{
    long long i = (long long)blockIdx.x * 256 + threadIdx.x;
    if (i < 16) { uint4v zz = {0u, 0u, 0u, 0u}; *(uint4v*)(zp + i * 8) = zz; }
    size_t base = (size_t)i * 8;
    if (base >= XBF_ELEMS) return;
    float4v f0 = *(const float4v*)(x + base);
    float4v f1 = *(const float4v*)(x + base + 4);
    uint4v u;
    u.x = bf16pair(f0.x, f0.y); u.y = bf16pair(f0.z, f0.w);
    u.z = bf16pair(f1.x, f1.y); u.w = bf16pair(f1.z, f1.w);
    *(uint4v*)(xbf + base) = u;
}

// -----------------------------------------------------------------------------
// Phase 2 (primary): implicit-GEMM conv, bf16 x source, double-buffered
// GLOBAL_LOAD_ASYNC_TO_LDS_B128 window staging overlapped with WMMA compute.
// One block per (b,z,y) row: M = 48 (3 sub-tiles), N = 128 (8 waves x 1 sub-tile).
// -----------------------------------------------------------------------------
__global__ __launch_bounds__(256) void voxconv_wmma_async(
    const unsigned short* __restrict__ xbf,   // bf16 x
    const unsigned short* __restrict__ Bbuf,  // fragment-ordered kernel
    const unsigned short* __restrict__ zp,    // 256B zero page (OOB source)
    float* __restrict__ out)
{
    __shared__ unsigned short lds[2 * 52 * LDSW];   // two 14,144B windows

    const int tid  = threadIdx.x;
    const int lane = tid & 31;
    const int nt   = tid >> 5;
    const int l15  = lane & 15;
    const int hi   = lane >> 4;

    const int y = blockIdx.x;
    const int z = blockIdx.y;
    const int b = blockIdx.z;

    const unsigned ldsbase = (unsigned)(size_t)(void*)lds;   // flat addr[31:0] == LDS byte offset

    // chs 120..127 pad (16B per slot) is never DMA'd: zero it once for both buffers
    for (int q = tid; q < 52 * 2; q += 256) {
        int bsel = q / 52, s = q % 52;
        uint4v zz = {0u, 0u, 0u, 0u};
        *(uint4v*)&lds[bsel * (52 * LDSW) + s * LDSW + 120] = zz;
    }

    // issue one window's 780 async 16B copies (chs 0..119 of 52 positions)
    auto issue = [&](int w, int bufsel) {
        const int dz = w / 5, dy = w % 5;
        const int zi = z + dz - 2, yi = y + dy - 2;
        const bool rowok = ((unsigned)zi < 48u) & ((unsigned)yi < 48u);
        const unsigned short* rowp =
            xbf + (size_t)(((b * 48 + (rowok ? zi : 0)) * 48 + (rowok ? yi : 0)) * 48) * DIM;
        const unsigned lbase = ldsbase + (unsigned)bufsel * (52 * LDSW * 2);
        for (int q = tid; q < 52 * 15; q += 256) {
            const int s = q / 15, c = q - s * 15;
            const int xg = s - 2;
            const bool v = rowok & ((unsigned)xg < 48u);
            const unsigned short* g = v ? (rowp + xg * DIM + c * 8) : zp;
            const unsigned laddr = lbase + (unsigned)(s * 272 + c * 16);
            const unsigned long long ga = (unsigned long long)(size_t)g;
            asm volatile("global_load_async_to_lds_b128 %0, %1, off"
                         :: "v"(laddr), "v"(ga) : "memory");
        }
    };

    v8f acc[3] = {};
    const unsigned short* bwave = Bbuf + nt * 512 + lane * 16;   // per-wave/lane B base

    issue(0, 0);
    for (int w = 0; w < 25; ++w) {
        const int cur = w & 1;
        if (w < 24) {
            issue(w + 1, cur ^ 1);   // prefetch next window (its buffer was drained last iter)
            // in-order async completion: <=3 outstanding => all of window w landed
            asm volatile("s_wait_asynccnt 0x3" ::: "memory");
        } else {
            asm volatile("s_wait_asynccnt 0x0" ::: "memory");
        }
        __syncthreads();             // window w visible to all waves

        const unsigned short* win = &lds[cur * (52 * LDSW)];
        #pragma unroll
        for (int dx = 0; dx < 5; ++dx) {
            const int t = w * 5 + dx;
            #pragma unroll
            for (int kt = 0; kt < 4; ++kt) {
                union { uint4v u[2]; v16bf v; } B;
                const uint4v* bp = (const uint4v*)(bwave + ((size_t)t * 16384 + kt * 4096));
                B.u[0] = bp[0];
                B.u[1] = bp[1];
                const int chb = kt * 32 + hi * 8;
                #pragma unroll
                for (int mt = 0; mt < 3; ++mt) {
                    union { uint4v u[2]; v16bf v; } A;
                    const unsigned short* ap = win + (mt * 16 + l15 + dx) * LDSW + chb;
                    A.u[0] = *(const uint4v*)ap;
                    A.u[1] = *(const uint4v*)(ap + 16);
                    acc[mt] = __builtin_amdgcn_wmma_f32_16x16x32_bf16(
                        false, A.v, false, B.v, (short)0, acc[mt], false, false);
                }
            }
        }
        __syncthreads();             // window w drained before its buffer is re-filled
    }

    const int orow = nt * 16 + l15;
    if (orow < DIM) {
        #pragma unroll
        for (int mt = 0; mt < 3; ++mt) {
            #pragma unroll
            for (int r = 0; r < 8; ++r) {
                const int m = mt * 16 + r + hi * 8;
                out[(size_t)(((b * 48 + z) * 48 + y) * 48 + m) * DIM + orow] = acc[mt][r];
            }
        }
    }
}

// -----------------------------------------------------------------------------
// Phase 2 (fallback, small ws): fp32 source, VALU-converted staging (round-1).
// -----------------------------------------------------------------------------
__global__ __launch_bounds__(256) void voxconv_wmma_sync(
    const float* __restrict__ x,
    const unsigned short* __restrict__ Bbuf,
    float* __restrict__ out)
{
    __shared__ unsigned short lds[52 * LDSW];

    const int tid  = threadIdx.x;
    const int lane = tid & 31;
    const int nt   = tid >> 5;
    const int l15  = lane & 15;
    const int hi   = lane >> 4;

    const int y = blockIdx.x;
    const int z = blockIdx.y;
    const int b = blockIdx.z;

    v8f acc[3] = {};
    const unsigned short* bwave = Bbuf + nt * 512 + lane * 16;

    for (int dz = 0; dz < 5; ++dz) {
        const int zi = z + dz - 2;
        for (int dy = 0; dy < 5; ++dy) {
            const int yi = y + dy - 2;
            const bool rowok = ((unsigned)zi < 48u) & ((unsigned)yi < 48u);
            const float* rowp =
                x + (size_t)(((b * 48 + (rowok ? zi : 0)) * 48 + (rowok ? yi : 0)) * 48) * DIM;

            __syncthreads();
            for (int q = tid; q < 52 * 32; q += 256) {
                const int s = q >> 5, cq = q & 31;
                const int xg = s - 2;
                unsigned int u01 = 0u, u23 = 0u;
                if (rowok && xg >= 0 && xg < 48 && cq < 30) {
                    const float4v f = *(const float4v*)(rowp + xg * DIM + cq * 4);
                    u01 = bf16pair(f.x, f.y);
                    u23 = bf16pair(f.z, f.w);
                }
                unsigned int* dst = (unsigned int*)&lds[s * LDSW + cq * 4];
                dst[0] = u01; dst[1] = u23;
            }
            __syncthreads();

            #pragma unroll
            for (int dx = 0; dx < 5; ++dx) {
                const int t = (dz * 5 + dy) * 5 + dx;
                #pragma unroll
                for (int kt = 0; kt < 4; ++kt) {
                    union { uint4v u[2]; v16bf v; } B;
                    const uint4v* bp = (const uint4v*)(bwave + ((size_t)t * 16384 + kt * 4096));
                    B.u[0] = bp[0];
                    B.u[1] = bp[1];
                    const int chb = kt * 32 + hi * 8;
                    #pragma unroll
                    for (int mt = 0; mt < 3; ++mt) {
                        union { uint4v u[2]; v16bf v; } A;
                        const unsigned short* ap = &lds[(mt * 16 + l15 + dx) * LDSW + chb];
                        A.u[0] = *(const uint4v*)ap;
                        A.u[1] = *(const uint4v*)(ap + 16);
                        acc[mt] = __builtin_amdgcn_wmma_f32_16x16x32_bf16(
                            false, A.v, false, B.v, (short)0, acc[mt], false, false);
                    }
                }
            }
        }
    }

    const int orow = nt * 16 + l15;
    if (orow < DIM) {
        #pragma unroll
        for (int mt = 0; mt < 3; ++mt) {
            #pragma unroll
            for (int r = 0; r < 8; ++r) {
                const int m = mt * 16 + r + hi * 8;
                out[(size_t)(((b * 48 + z) * 48 + y) * 48 + m) * DIM + orow] = acc[mt][r];
            }
        }
    }
}

extern "C" void kernel_launch(void* const* d_in, const int* in_sizes, int n_in,
                              void* d_out, int out_size, void* d_ws, size_t ws_size,
                              hipStream_t stream) {
    const float* x   = (const float*)d_in[0];   // (4,48,48,48,120)
    const float* wgt = (const float*)d_in[1];   // (3,5184)
    const float* scw = (const float*)d_in[2];   // (2880,)
    float* out = (float*)d_out;
    unsigned short* Bbuf = (unsigned short*)d_ws;

    const int total = NLAT * 16384;
    build_wmma_kernel<<<(total + 255) / 256, 256, 0, stream>>>(wgt, scw, Bbuf);

    if (ws_size >= WS_NEED) {
        unsigned short* xbf = (unsigned short*)((char*)d_ws + KBUF_BYTES);
        unsigned short* zp  = (unsigned short*)((char*)d_ws + ZP_OFF);
        const long long nconv = (long long)(XBF_ELEMS / 8);          // 6,635,520 threads
        convert_bf16<<<(unsigned)((nconv + 255) / 256), 256, 0, stream>>>(x, xbf, zp);
        voxconv_wmma_async<<<dim3(48, 48, 4), 256, 0, stream>>>(xbf, Bbuf, zp, out);
    } else {
        voxconv_wmma_sync<<<dim3(48, 48, 4), 256, 0, stream>>>(x, Bbuf, out);
    }
}
